// RNNModel_32126355374137
// MI455X (gfx1250) — matrix operand order repaired and checked by reading, not compiled, not run on hardware
//
#include <hip/hip_runtime.h>

typedef __attribute__((ext_vector_type(16))) _Float16 v16h;
typedef __attribute__((ext_vector_type(8)))  float    v8f;

static constexpr int HDIM  = 64;    // hidden size
static constexpr int G4H   = 256;   // 4*H gate width
static constexpr int BTILE = 16;    // batch rows per workgroup (WMMA M)
static constexpr int TLEN  = 2048;
static constexpr int BATCH = 512;

static __device__ inline v8f wmma_f16(v16h a, v16h b, v8f c) {
  // D = A(16x32 f16) * B(32x16 f16) + C(16x16 f32)
  return __builtin_amdgcn_wmma_f32_16x16x32_f16(
      /*neg_a=*/false, a, /*neg_b=*/false, b,
      /*c_mod=*/(short)0, c, /*reuse_a=*/false, /*reuse_b=*/false);
}

// Native v_tanh_f32 on CDNA5 (single TRANS op, co-executes with WMMA);
// branch-free fallback otherwise.
static __device__ inline float fast_tanh(float x) {
#if __has_builtin(__builtin_amdgcn_tanhf)
  return __builtin_amdgcn_tanhf(x);
#else
  float ax = fabsf(x);
  float e  = __expf(2.0f * ax);
  float t  = 1.0f - 2.0f / (e + 1.0f);
  return copysignf(t, x);
#endif
}
static __device__ inline float sigm(float x) {
  return 0.5f + 0.5f * fast_tanh(0.5f * x);
}

// Build A-fragment (16x32 f16) from an LDS h-tile stored row-major [16][64].
// lanes 0-15 hold M=lane%16; lane-half selects contiguous K block of 16 (code's own order, B matches).
static __device__ inline v16h build_a(const _Float16* __restrict__ ht,
                                      int lane, int kb) {
  const int M  = lane & 15;
  const int hi = lane >> 4;
  v16h a;
#pragma unroll
  for (int e = 0; e < 16; ++e) {
    const int K = kb * 32 + 16 * hi + e;
    a[e] = ht[M * HDIM + K];
  }
  return a;
}

__global__ __launch_bounds__(256)
void lstm2_pipelined_kernel(const float* __restrict__ x,
                            const float* __restrict__ w_ih0,
                            const float* __restrict__ w_hh0,
                            const float* __restrict__ b_ih0,
                            const float* __restrict__ b_hh0,
                            const float* __restrict__ w_ih1,
                            const float* __restrict__ w_hh1,
                            const float* __restrict__ b_ih1,
                            const float* __restrict__ b_hh1,
                            const float* __restrict__ fc1_w,
                            const float* __restrict__ fc1_b,
                            const float* __restrict__ fc2_w,
                            const float* __restrict__ fc2_b,
                            float* __restrict__ out)
{
  // 32KB buffer reused as: f16 weight staging -> fc1 activations
  __shared__ __align__(16) unsigned char smemRaw[32768];
  __shared__ _Float16 h1buf[2][BTILE * HDIM];  // double-buffered layer-1 hidden
  __shared__ _Float16 h2buf[2][BTILE * HDIM];  // double-buffered layer-2 hidden
  __shared__ float    h2f[BTILE * HDIM];       // final hidden f32 (for head)
  __shared__ float    bias0[G4H];
  __shared__ float    bias1[G4H];
  __shared__ float    wih0v[G4H];              // w_ih0 is [4H,1] -> vector
  __shared__ float    xbuf[2][BTILE];          // double-buffered x column

  _Float16* stage = (_Float16*)smemRaw;        // 256*64 f16 = 32KB

  const int tid   = threadIdx.x;
  const int lane  = tid & 31;
  const int wave  = tid >> 5;                  // 8 waves
  const int grp   = wave >> 2;                 // 0 = layer-1 group, 1 = layer-2
  const int g     = wave & 3;                  // j-tile (16 hidden cols) in group
  const int lhi   = lane >> 4;
  const int laneN = lane & 15;
  const int b0    = blockIdx.x * BTILE;

  // ---- init state + small tables ----
  for (int i = tid; i < BTILE * HDIM; i += 256) {
    h1buf[0][i] = (_Float16)0.0f; h1buf[1][i] = (_Float16)0.0f;
    h2buf[0][i] = (_Float16)0.0f; h2buf[1][i] = (_Float16)0.0f;
  }
  for (int i = tid; i < G4H; i += 256) {
    bias0[i] = b_ih0[i] + b_hh0[i];
    bias1[i] = b_ih1[i] + b_hh1[i];
    wih0v[i] = w_ih0[i];
  }

  // ---- one-time: gather weight B-fragments into VGPRs ----
  // Wave owns j-columns [16g,16g+16): gate tiles nt = g, g+4, g+8, g+12
  // (i,f,g,o) so the per-cell update is fully in-lane, in-register.
  // Group A: fA = w_hh0.  Group B: fA = w_ih1, fB = w_hh1.
  v16h fA[4][2], fB[4][2];
  const float* const wsrc[3] = { w_hh0, w_ih1, w_hh1 };
#pragma unroll 1
  for (int m = 0; m < 3; ++m) {
    __syncthreads();
    for (int i = tid; i < G4H * HDIM; i += 256)
      stage[i] = (_Float16)wsrc[m][i];         // [N][K] row-major, f32->f16
    __syncthreads();
#pragma unroll
    for (int q = 0; q < 4; ++q) {
      const int N = (g + 4 * q) * 16 + laneN;
#pragma unroll
      for (int kb = 0; kb < 2; ++kb) {
        const int kbase = kb * 32 + 16 * lhi;
        v16h f;
#pragma unroll
        for (int e = 0; e < 16; ++e) f[e] = stage[N * HDIM + kbase + e];
        if (m == 0 && grp == 0) fA[q][kb] = f;
        if (m == 1 && grp == 1) fA[q][kb] = f;
        if (m == 2 && grp == 1) fB[q][kb] = f;
      }
    }
  }

  // cell state lives in registers for the whole scan
  v8f c1, c2;
#pragma unroll
  for (int r = 0; r < 8; ++r) { c1[r] = 0.0f; c2[r] = 0.0f; }

  if (tid < BTILE) xbuf[0][tid] = x[(size_t)(b0 + tid) * TLEN + 0];
  __syncthreads();

  // ---- pipelined scan: group A does layer-1 step k, group B layer-2 step k-1
#pragma unroll 1
  for (int k = 0; k <= TLEN; ++k) {
    const int cur = k & 1;
    const int prv = cur ^ 1;  // (k-1)&1

    if (grp == 0) {
      if (k < TLEN) {
        // layer 1: h1[k] = cell(x[k]*w_ih0 + bias0 + h1[k-1] @ w_hh0^T)
        v16h a0 = build_a(h1buf[prv], lane, 0);
        v16h a1 = build_a(h1buf[prv], lane, 1);
        v8f acc[4];
#pragma unroll
        for (int q = 0; q < 4; ++q) {
          const int N = (g + 4 * q) * 16 + laneN;
#pragma unroll
          for (int r = 0; r < 8; ++r)
            acc[q][r] = xbuf[cur][r + 8 * lhi] * wih0v[N] + bias0[N];
          acc[q] = wmma_f16(a0, fA[q][0], acc[q]);
          acc[q] = wmma_f16(a1, fA[q][1], acc[q]);
        }
#pragma unroll
        for (int r = 0; r < 8; ++r) {
          const float i_ = sigm(acc[0][r]);
          const float f_ = sigm(acc[1][r]);
          const float g_ = fast_tanh(acc[2][r]);
          const float o_ = sigm(acc[3][r]);
          const float c  = f_ * c1[r] + i_ * g_;
          c1[r] = c;
          const float h = o_ * fast_tanh(c);
          h1buf[cur][(r + 8 * lhi) * HDIM + 16 * g + laneN] = (_Float16)h;
        }
        // prefetch next x column (no extra barrier needed)
        if (wave == 0 && lane < BTILE && (k + 1) < TLEN)
          xbuf[prv][lane] = x[(size_t)(b0 + lane) * TLEN + (k + 1)];
      }
    } else {
      if (k >= 1) {
        // layer 2 @ step k-1: gates = bias1 + h1[k-1]@w_ih1^T + h2[k-2]@w_hh1^T
        v16h p0 = build_a(h1buf[prv], lane, 0);
        v16h p1 = build_a(h1buf[prv], lane, 1);
        v16h q0 = build_a(h2buf[cur], lane, 0);   // h2[k-2]
        v16h q1 = build_a(h2buf[cur], lane, 1);
        v8f acc[4];
#pragma unroll
        for (int q = 0; q < 4; ++q) {
          const int N = (g + 4 * q) * 16 + laneN;
#pragma unroll
          for (int r = 0; r < 8; ++r) acc[q][r] = bias1[N];
          acc[q] = wmma_f16(p0, fA[q][0], acc[q]);
          acc[q] = wmma_f16(p1, fA[q][1], acc[q]);
          acc[q] = wmma_f16(q0, fB[q][0], acc[q]);
          acc[q] = wmma_f16(q1, fB[q][1], acc[q]);
        }
#pragma unroll
        for (int r = 0; r < 8; ++r) {
          const float i_ = sigm(acc[0][r]);
          const float f_ = sigm(acc[1][r]);
          const float g_ = fast_tanh(acc[2][r]);
          const float o_ = sigm(acc[3][r]);
          const float c  = f_ * c2[r] + i_ * g_;
          c2[r] = c;
          const float h = o_ * fast_tanh(c);
          const int idx = (r + 8 * lhi) * HDIM + 16 * g + laneN;
          h2buf[prv][idx] = (_Float16)h;          // h2[k-1]
          if (k == TLEN) h2f[idx] = h;            // final hidden, f32
        }
      }
    }
    __syncthreads();   // single barrier per pipelined step
  }

  // ===== MLP head on final h2: relu(hT@fc1^T+b1) @ fc2^T + b2 =====
  float* out1 = (float*)smemRaw;                  // reuse smem: 16*128 f32
#pragma unroll 1
  for (int q = 0; q < 8; ++q) {
    const int idx = tid + q * 256;                // 0..2047 == m*128 + n
    const int m = idx >> 7, n = idx & 127;
    float acc = fc1_b[n];
    for (int kk = 0; kk < HDIM; ++kk)
      acc += h2f[m * HDIM + kk] * fc1_w[n * HDIM + kk];
    out1[idx] = fmaxf(acc, 0.0f);
  }
  __syncthreads();
  if (tid < BTILE * 10) {
    const int m = tid / 10, p = tid % 10;
    float acc = fc2_b[p];
    for (int n = 0; n < 128; ++n) acc += out1[m * 128 + n] * fc2_w[p * 128 + n];
    out[(size_t)(b0 + m) * 10 + p] = acc;
  }
}

extern "C" void kernel_launch(void* const* d_in, const int* in_sizes, int n_in,
                              void* d_out, int out_size, void* d_ws, size_t ws_size,
                              hipStream_t stream) {
  (void)in_sizes; (void)n_in; (void)out_size; (void)d_ws; (void)ws_size;
  const float* x     = (const float*)d_in[0];
  const float* w_ih0 = (const float*)d_in[1];
  const float* w_hh0 = (const float*)d_in[2];
  const float* b_ih0 = (const float*)d_in[3];
  const float* b_hh0 = (const float*)d_in[4];
  const float* w_ih1 = (const float*)d_in[5];
  const float* w_hh1 = (const float*)d_in[6];
  const float* b_ih1 = (const float*)d_in[7];
  const float* b_hh1 = (const float*)d_in[8];
  const float* fc1_w = (const float*)d_in[9];
  const float* fc1_b = (const float*)d_in[10];
  const float* fc2_w = (const float*)d_in[11];
  const float* fc2_b = (const float*)d_in[12];
  float* out = (float*)d_out;

  dim3 grid(BATCH / BTILE);   // 32 workgroups (one 16-row batch tile each)
  dim3 block(256);            // 8 wave32 waves: 4 layer-1 + 4 layer-2
  lstm2_pipelined_kernel<<<grid, block, 0, stream>>>(
      x, w_ih0, w_hh0, b_ih0, b_hh0, w_ih1, w_hh1, b_ih1, b_hh1,
      fc1_w, fc1_b, fc2_w, fc2_b, out);
}
